// PreAttention_15032385536187
// MI455X (gfx1250) — compile-verified, matrix operand order checked by reading, not run on hardware
//
#include <hip/hip_runtime.h>
#include <hip/hip_bf16.h>
#include <math.h>

// ---------- types for WMMA ----------
typedef __attribute__((ext_vector_type(16))) __bf16 v16bf;
typedef __attribute__((ext_vector_type(8)))  __bf16 v8bf;
typedef __attribute__((ext_vector_type(8)))  float  v8f;

union Frag { v16bf v; v8bf h[2]; };

#define BM 128
#define BN 128
#define BK 64
#define LDSP (BK + 8)   // 72 bf16 = 144B row stride -> conflict-free b128 LDS reads

// ---- CDNA5 async global->LDS path (ASYNCcnt) ----
#if defined(__has_builtin)
# if __has_builtin(__builtin_amdgcn_global_load_async_to_lds_b128)
#  define USE_ASYNC 1
# endif
#endif
#ifndef USE_ASYNC
# define USE_ASYNC 0
#endif

typedef int b128v __attribute__((vector_size(16)));   // V4i, matches builtin param type
typedef __attribute__((address_space(1))) b128v* g128p;
typedef __attribute__((address_space(3))) b128v* l128p;

__device__ __forceinline__ void async_b128(const __bf16* g, __bf16* l) {
#if USE_ASYNC
    __builtin_amdgcn_global_load_async_to_lds_b128(
        (g128p)(uintptr_t)g,
        (l128p)(uint32_t)(uintptr_t)l,    // flat LDS addr truncates to as3 offset (ISA aperture rule)
        0, 0);
#else
    (void)g; (void)l;
#endif
}

__device__ __forceinline__ void wait_async0() {
#if USE_ASYNC
# if __has_builtin(__builtin_amdgcn_s_wait_asynccnt)
    __builtin_amdgcn_s_wait_asynccnt(0);
# else
    asm volatile("s_wait_asynccnt 0" ::: "memory");
# endif
#endif
}

// =====================================================================
// GEMM: C[M,N] = scale * (A[M,K] @ Bt[N,K]^T) + bias
// A: bf16 row-major [M,K]; Bt: bf16 row-major [N,K] (pre-transposed weight)
// grid = (N/128, M/128), block = 256 threads = 8 waves (wave32)
// double-buffered LDS, one barrier per 64-deep K-step, 16 wmma per barrier
// =====================================================================
__global__ __launch_bounds__(256)
void gemm_bf16_kernel(const __bf16* __restrict__ A, const __bf16* __restrict__ Bt,
                      const float* __restrict__ bias, const float* __restrict__ scale_ptr,
                      float* __restrict__ C, int M, int N, int K) {
    __shared__ __bf16 As[2][BM][LDSP];
    __shared__ __bf16 Bs[2][BN][LDSP];

    const int tid  = threadIdx.x;
    const int lane = tid & 31;
    const int wave = tid >> 5;

    const int bm = blockIdx.y * BM;
    const int bn = blockIdx.x * BN;
    const int wm = (wave >> 1) * 32;        // 4 wave rows
    const int wn = (wave & 1)  * 64;        // 2 wave cols
    const int lr    = lane & 15;
    const int khalf = (lane < 16) ? 0 : 8;  // 16-bit fragment K layout

    // tile fill map (identical for As and Bs): 128 rows x 64 cols bf16, 32 per thread
    const int fr = tid >> 1;
    const int fc = (tid & 1) * 32;
    const __bf16* gaRow = A  + (long)(bm + fr) * K + fc;
    const __bf16* gbRow = Bt + (long)(bn + fr) * K + fc;

    const int nit = K / BK;

    v8f acc[2][4];
    #pragma unroll
    for (int mt = 0; mt < 2; mt++)
        #pragma unroll
        for (int nt = 0; nt < 4; nt++) acc[mt][nt] = v8f{};

#if USE_ASYNC
    // prologue: async-fill buffer 0
    #pragma unroll
    for (int c = 0; c < 4; c++) {
        async_b128(gaRow + 8 * c, &As[0][fr][fc + 8 * c]);
        async_b128(gbRow + 8 * c, &Bs[0][fr][fc + 8 * c]);
    }
    wait_async0();
    __syncthreads();
#else
    uint4 ra[4], rb[4];
    #pragma unroll
    for (int c = 0; c < 4; c++) {
        ra[c] = *(const uint4*)(gaRow + 8 * c);
        rb[c] = *(const uint4*)(gbRow + 8 * c);
    }
    #pragma unroll
    for (int c = 0; c < 4; c++) {
        *(uint4*)&As[0][fr][fc + 8 * c] = ra[c];
        *(uint4*)&Bs[0][fr][fc + 8 * c] = rb[c];
    }
    __syncthreads();
#endif

    for (int it = 0; it < nit; it++) {
        const int buf  = it & 1;
        const bool more = (it + 1) < nit;
        const int konext = (it + 1) * BK;

#if USE_ASYNC
        if (more) {
            #pragma unroll
            for (int c = 0; c < 4; c++) {
                async_b128(gaRow + konext + 8 * c, &As[buf ^ 1][fr][fc + 8 * c]);
                async_b128(gbRow + konext + 8 * c, &Bs[buf ^ 1][fr][fc + 8 * c]);
            }
        }
#else
        if (more) {
            #pragma unroll
            for (int c = 0; c < 4; c++) {
                ra[c] = *(const uint4*)(gaRow + konext + 8 * c);
                rb[c] = *(const uint4*)(gbRow + konext + 8 * c);
            }
            __builtin_prefetch(gaRow + konext + BK, 0, 1);
            __builtin_prefetch(gbRow + konext + BK, 0, 1);
        }
#endif

        // compute on current buffer: 2 sub-K steps x 8 wmma
        #pragma unroll
        for (int kk = 0; kk < 2; kk++) {
            Frag a[2], b[4];
            #pragma unroll
            for (int mt = 0; mt < 2; mt++) {
                a[mt].h[0] = *(const v8bf*)&As[buf][wm + mt * 16 + lr][kk * 32 + khalf];
                a[mt].h[1] = *(const v8bf*)&As[buf][wm + mt * 16 + lr][kk * 32 + 16 + khalf];
            }
            #pragma unroll
            for (int nt = 0; nt < 4; nt++) {
                b[nt].h[0] = *(const v8bf*)&Bs[buf][wn + nt * 16 + lr][kk * 32 + khalf];
                b[nt].h[1] = *(const v8bf*)&Bs[buf][wn + nt * 16 + lr][kk * 32 + 16 + khalf];
            }
            #pragma unroll
            for (int mt = 0; mt < 2; mt++)
                #pragma unroll
                for (int nt = 0; nt < 4; nt++)
                    acc[mt][nt] = __builtin_amdgcn_wmma_f32_16x16x32_bf16(
                        false, a[mt].v, false, b[nt].v, (short)0, acc[mt][nt], false, false);
        }

#if USE_ASYNC
        if (more) wait_async0();
#else
        if (more) {
            #pragma unroll
            for (int c = 0; c < 4; c++) {
                *(uint4*)&As[buf ^ 1][fr][fc + 8 * c] = ra[c];
                *(uint4*)&Bs[buf ^ 1][fr][fc + 8 * c] = rb[c];
            }
        }
#endif
        __syncthreads();
    }

    const float scl  = scale_ptr ? scale_ptr[0] : 1.0f;
    const int   rofs = (lane < 16) ? 0 : 8;   // f32 C/D layout: VGPR r -> M=r (lo lanes) / r+8 (hi)

    #pragma unroll
    for (int mt = 0; mt < 2; mt++) {
        #pragma unroll
        for (int nt = 0; nt < 4; nt++) {
            int row0 = bm + wm + mt * 16 + rofs;
            int col  = bn + wn + nt * 16 + lr;
            float bv = bias ? bias[col] : 0.0f;
            #pragma unroll
            for (int r = 0; r < 8; r++)
                C[(long)(row0 + r) * N + col] = acc[mt][nt][r] * scl + bv;
        }
    }
}

// =====================================================================
// f32 -> bf16 elementwise (8 elems / thread)
// =====================================================================
__global__ __launch_bounds__(256)
void convert_bf16_kernel(const float* __restrict__ in, __bf16* __restrict__ out, long n) {
    long i = ((long)blockIdx.x * blockDim.x + threadIdx.x) * 8;
    if (i < n) {
        float4 a = *(const float4*)(in + i);
        float4 b = *(const float4*)(in + i + 4);
        v8bf o;
        o[0]=(__bf16)a.x; o[1]=(__bf16)a.y; o[2]=(__bf16)a.z; o[3]=(__bf16)a.w;
        o[4]=(__bf16)b.x; o[5]=(__bf16)b.y; o[6]=(__bf16)b.z; o[7]=(__bf16)b.w;
        *(v8bf*)(out + i) = o;
    }
}

// =====================================================================
// W[K,N] f32 -> Wt[N,K] bf16 (32x32 LDS tile transpose)
// grid = (N/32, K/32), block = 256
// =====================================================================
__global__ __launch_bounds__(256)
void transpose_bf16_kernel(const float* __restrict__ W, __bf16* __restrict__ Wt, int K, int N) {
    __shared__ __bf16 t[32][33];
    const int k0 = blockIdx.y * 32, n0 = blockIdx.x * 32;
    for (int i = threadIdx.x; i < 1024; i += 256) {
        int r = i >> 5, c = i & 31;
        t[r][c] = (__bf16)W[(long)(k0 + r) * N + n0 + c];
    }
    __syncthreads();
    for (int i = threadIdx.x; i < 1024; i += 256) {
        int r = i >> 5, c = i & 31;
        Wt[(long)(n0 + r) * K + k0 + c] = t[c][r];
    }
}

// =====================================================================
// Attention combine: per group of 8 rows (fixed b,h):
//   a[w][v] = sigmoid(dot(f[w,:], g[v,:]));  out[w,c] = sum_v a[w][v]*h[v,c] + res[w,c]
// writes bf16 (feeds the next GEMM). grid = M/8, block = 256
// =====================================================================
__global__ __launch_bounds__(256)
void combine_kernel(const float* __restrict__ f, const float* __restrict__ g,
                    const float* __restrict__ h, const float* __restrict__ res,
                    __bf16* __restrict__ out, int Nf, int C) {
    __shared__ float sf[8 * 256];
    __shared__ float sg[8 * 256];
    __shared__ float sa[8][8];
    const long base = (long)blockIdx.x * 8;

    for (int i = threadIdx.x; i < 8 * Nf; i += blockDim.x) {
        int w = i / Nf, k = i - w * Nf;
        sf[w * Nf + k] = f[(base + w) * Nf + k];
        sg[w * Nf + k] = g[(base + w) * Nf + k];
    }
    __syncthreads();
    if (threadIdx.x < 64) {
        int w = threadIdx.x >> 3, v = threadIdx.x & 7;
        float s = 0.f;
        for (int k = 0; k < Nf; k++) s += sf[w * Nf + k] * sg[v * Nf + k];
        sa[w][v] = 1.0f / (1.0f + __expf(-s));
    }
    __syncthreads();

    for (int c = threadIdx.x; c < C; c += blockDim.x) {
        float hv[8];
        #pragma unroll
        for (int v = 0; v < 8; v++) hv[v] = h[(base + v) * C + c];
        #pragma unroll
        for (int w = 0; w < 8; w++) {
            float s = res[(base + w) * C + c];
            #pragma unroll
            for (int v = 0; v < 8; v++) s += sa[w][v] * hv[v];
            out[(base + w) * C + c] = (__bf16)s;
        }
    }
}

// =====================================================================
// out = layernorm(relu(in)) * gamma + beta   (one workgroup per row)
// optionally also writes a bf16 copy (feeds next GEMM)
// =====================================================================
__global__ __launch_bounds__(256)
void relu_ln_kernel(const float* __restrict__ in, const float* __restrict__ gamma,
                    const float* __restrict__ beta, float* __restrict__ out,
                    __bf16* __restrict__ out_bf, int N) {
    const long row = blockIdx.x;
    const float* p = in + row * N;
    __shared__ float rs[256], rq[256];
    float ls = 0.f, lq = 0.f;
    for (int c = threadIdx.x; c < N; c += blockDim.x) {
        float v = fmaxf(p[c], 0.f);
        ls += v; lq += v * v;
    }
    rs[threadIdx.x] = ls; rq[threadIdx.x] = lq;
    __syncthreads();
    for (int o = 128; o > 0; o >>= 1) {
        if (threadIdx.x < o) { rs[threadIdx.x] += rs[threadIdx.x + o]; rq[threadIdx.x] += rq[threadIdx.x + o]; }
        __syncthreads();
    }
    float mean = rs[0] / (float)N;
    float var  = rq[0] / (float)N - mean * mean;
    float inv  = rsqrtf(var + 1e-6f);
    for (int c = threadIdx.x; c < N; c += blockDim.x) {
        float v  = fmaxf(p[c], 0.f);
        float o2 = (v - mean) * inv * gamma[c] + beta[c];
        if (out)    out[row * N + c]    = o2;
        if (out_bf) out_bf[row * N + c] = (__bf16)o2;
    }
}

// =====================================================================
// Spectral norm: t = u@W^T ; v = t/(||t||+eps) ; s = v@W ;
// sigma = ||s||^2/(||s||+eps) ; writes 1/sigma. Single workgroup.
// =====================================================================
__global__ __launch_bounds__(256)
void sigma_kernel(const float* __restrict__ W, const float* __restrict__ u,
                  int cin, int cout, float* __restrict__ inv_sigma) {
    __shared__ float t[2048];
    __shared__ float red[256];
    const int tid = threadIdx.x;

    float ls = 0.f;
    for (int i = tid; i < cin; i += 256) {
        float s = 0.f;
        for (int j = 0; j < cout; j++) s += u[j] * W[(long)i * cout + j];
        t[i] = s; ls += s * s;
    }
    red[tid] = ls; __syncthreads();
    for (int o = 128; o > 0; o >>= 1) { if (tid < o) red[tid] += red[tid + o]; __syncthreads(); }
    float tinv = 1.0f / (sqrtf(red[0]) + 1e-12f);
    __syncthreads();

    float lq = 0.f;
    for (int j = tid; j < cout; j += 256) {
        float s = 0.f;
        for (int i = 0; i < cin; i++) s += t[i] * W[(long)i * cout + j];
        s *= tinv;
        lq += s * s;
    }
    red[tid] = lq; __syncthreads();
    for (int o = 128; o > 0; o >>= 1) { if (tid < o) red[tid] += red[tid + o]; __syncthreads(); }
    if (tid == 0) {
        float ssq = red[0];
        float sig = ssq / (sqrtf(ssq) + 1e-12f);
        inv_sigma[0] = 1.0f / sig;
    }
}

// =====================================================================
extern "C" void kernel_launch(void* const* d_in, const int* in_sizes, int n_in,
                              void* d_out, int out_size, void* d_ws, size_t ws_size,
                              hipStream_t stream) {
    const float* x    = (const float*)d_in[0];
    const float* f1w  = (const float*)d_in[1];
    const float* f1b  = (const float*)d_in[2];
    const float* g1w  = (const float*)d_in[3];
    const float* g1b  = (const float*)d_in[4];
    const float* h1w  = (const float*)d_in[5];
    const float* h1b  = (const float*)d_in[6];
    const float* w1   = (const float*)d_in[7];
    const float* u1   = (const float*)d_in[8];
    const float* ln1g = (const float*)d_in[9];
    const float* ln1b = (const float*)d_in[10];
    const float* f2w  = (const float*)d_in[11];
    const float* f2b  = (const float*)d_in[12];
    const float* g2w  = (const float*)d_in[13];
    const float* g2b  = (const float*)d_in[14];
    const float* h2w  = (const float*)d_in[15];
    const float* h2b  = (const float*)d_in[16];
    const float* w2   = (const float*)d_in[17];
    const float* u2   = (const float*)d_in[18];
    const float* ln2g = (const float*)d_in[19];
    const float* ln2b = (const float*)d_in[20];

    const int M = 32768;                    // 512 * 8 * 8 rows
    char* ws = (char*)d_ws;

    // ---- workspace layout (liveness-based reuse), ~622 MB ----
    __bf16* xb    = (__bf16*)(ws + 0L);            // 134.2 MB  x as bf16 (later reused as x2 f32)
    float*  x2f   = (float*)(ws + 0L);             //           x2 f32 (after xb is dead)
    float*  wsA   = (float*)(ws + 134217728L);     //  33.5 MB  y_f1
    float*  wsB   = (float*)(ws + 167772160L);     //  33.5 MB  y_g1
    __bf16* x2bf  = (__bf16*)(ws + 134217728L);    //  67.1 MB  x2 bf16 (reuses wsA+wsB)
    float*  wsC   = (float*)(ws + 201326592L);     // 268.4 MB  y_h1 / y_sn1 / y_h2 / y_sn2
    __bf16* x1b   = (__bf16*)(ws + 469762048L);    // 134.2 MB  block1 out bf16
    float*  f2out = (float*)(ws + 469762048L);     //  16.8 MB  (reuses x1b after it's dead)
    float*  g2out = (float*)(ws + 486539264L);     //  16.8 MB
    __bf16* x2bb  = (__bf16*)(ws + 536870912L);    //  67.1 MB  block2 out bf16
    __bf16* wts   = (__bf16*)(ws + 603979776L);    //  18.4 MB  transposed bf16 weights
    float*  scal  = (float*)(ws + 622329856L);     //  inv_sigma1, inv_sigma2

    __bf16* f1t = wts + 0L;        // [256,2048]
    __bf16* g1t = wts + 524288L;   // [256,2048]
    __bf16* h1t = wts + 1048576L;  // [2048,2048]
    __bf16* w1t = wts + 5242880L;  // [1024,2048]
    __bf16* f2t = wts + 7340032L;  // [128,1024]
    __bf16* g2t = wts + 7471104L;  // [128,1024]
    __bf16* h2t = wts + 7602176L;  // [1024,1024]
    __bf16* w2t = wts + 8650752L;  // [512,1024]

    // ---- preprocessing: sigmas, bf16 conversions, weight transposes ----
    sigma_kernel<<<1, 256, 0, stream>>>(w1, u1, 2048, 1024, scal + 0);
    sigma_kernel<<<1, 256, 0, stream>>>(w2, u2, 1024, 512, scal + 1);

    convert_bf16_kernel<<<(long)M * 2048 / (256 * 8), 256, 0, stream>>>(x, xb, (long)M * 2048);

    transpose_bf16_kernel<<<dim3(256/32, 2048/32), 256, 0, stream>>>(f1w, f1t, 2048, 256);
    transpose_bf16_kernel<<<dim3(256/32, 2048/32), 256, 0, stream>>>(g1w, g1t, 2048, 256);
    transpose_bf16_kernel<<<dim3(2048/32, 2048/32), 256, 0, stream>>>(h1w, h1t, 2048, 2048);
    transpose_bf16_kernel<<<dim3(1024/32, 2048/32), 256, 0, stream>>>(w1, w1t, 2048, 1024);
    transpose_bf16_kernel<<<dim3(128/32, 1024/32), 256, 0, stream>>>(f2w, f2t, 1024, 128);
    transpose_bf16_kernel<<<dim3(128/32, 1024/32), 256, 0, stream>>>(g2w, g2t, 1024, 128);
    transpose_bf16_kernel<<<dim3(1024/32, 1024/32), 256, 0, stream>>>(h2w, h2t, 1024, 1024);
    transpose_bf16_kernel<<<dim3(512/32, 1024/32), 256, 0, stream>>>(w2, w2t, 1024, 512);

    // ---- block 1 ----
    gemm_bf16_kernel<<<dim3(256/BN, M/BM), 256, 0, stream>>>(xb, f1t, f1b, nullptr, wsA, M, 256, 2048);
    gemm_bf16_kernel<<<dim3(256/BN, M/BM), 256, 0, stream>>>(xb, g1t, g1b, nullptr, wsB, M, 256, 2048);
    gemm_bf16_kernel<<<dim3(2048/BN, M/BM), 256, 0, stream>>>(xb, h1t, h1b, nullptr, wsC, M, 2048, 2048);
    combine_kernel<<<M / 8, 256, 0, stream>>>(wsA, wsB, wsC, x, x1b, 256, 2048);

    // spectral conv 1 -> relu -> layernorm (writes f32 x2 + bf16 x2)
    gemm_bf16_kernel<<<dim3(1024/BN, M/BM), 256, 0, stream>>>(x1b, w1t, nullptr, scal + 0, wsC, M, 1024, 2048);
    relu_ln_kernel<<<M, 256, 0, stream>>>(wsC, ln1g, ln1b, x2f, x2bf, 1024);

    // ---- block 2 ----
    gemm_bf16_kernel<<<dim3(128/BN, M/BM), 256, 0, stream>>>(x2bf, f2t, f2b, nullptr, f2out, M, 128, 1024);
    gemm_bf16_kernel<<<dim3(128/BN, M/BM), 256, 0, stream>>>(x2bf, g2t, g2b, nullptr, g2out, M, 128, 1024);
    gemm_bf16_kernel<<<dim3(1024/BN, M/BM), 256, 0, stream>>>(x2bf, h2t, h2b, nullptr, wsC, M, 1024, 1024);
    combine_kernel<<<M / 8, 256, 0, stream>>>(f2out, g2out, wsC, x2f, x2bb, 128, 1024);

    // spectral conv 2 -> relu -> layernorm -> output
    gemm_bf16_kernel<<<dim3(512/BN, M/BM), 256, 0, stream>>>(x2bb, w2t, nullptr, scal + 1, wsC, M, 512, 1024);
    relu_ln_kernel<<<M, 256, 0, stream>>>(wsC, ln2g, ln2b, (float*)d_out, 512 ? nullptr : nullptr, 512);
}
